// GIBSAGE_59416577573179
// MI455X (gfx1250) — compile-verified, matrix operand order checked by reading, not run on hardware
//
#include <hip/hip_runtime.h>
#include <hip/hip_bf16.h>
#include <stdint.h>

// ---------------------------------------------------------------------------
// GIBSAGE pipeline for MI455X (gfx1250, wave32, WMMA).
// GEMMs: bf16 WMMA 16x16x32, f32 accumulate, A tiles staged in LDS (padded
// rows to avoid bank conflicts). Scatters: wave-per-edge f32 atomics.
// ---------------------------------------------------------------------------

typedef __bf16 v16bf __attribute__((ext_vector_type(16)));
typedef float  v8f   __attribute__((ext_vector_type(8)));

static constexpr int Nn = 100000;   // nodes
static constexpr int Hh = 128;      // F == H == 128
static constexpr int Ee = 640000;   // edges
static constexpr int Gg = 512;      // graphs
static constexpr int Cc = 6;        // classes

__device__ __forceinline__ unsigned short f32_to_bf16_rne(float f) {
  unsigned int u = __float_as_uint(f);
  if ((u & 0x7f800000u) == 0x7f800000u) return (unsigned short)(u >> 16); // inf/nan
  unsigned int r = u + 0x7fffu + ((u >> 16) & 1u);
  return (unsigned short)(r >> 16);
}

union FragBF { uint4 u[2]; v16bf v; };

// -------------------------- casts / packing --------------------------------

__global__ void cast_f32_to_bf16(const float* __restrict__ in,
                                 unsigned short* __restrict__ out, int n) {
  int i = blockIdx.x * blockDim.x + threadIdx.x;
  if (i < n) out[i] = f32_to_bf16_rne(in[i]);
}

// Pack a 128x128 f32 weight (K x N, row-major) into WMMA-B-native bf16 layout:
// packed[kt][nt][lane][p], p=0..15 contiguous per lane (one 32B load per frag).
// 16-bit B layout: lane holds col n = nt*16 + (lane&15); half h = lane>>4;
// element p -> k = kt*32 + (p<8 ? 8h+p : 16+8h+(p-8)).
__global__ void pack_weight_bf16(const float* __restrict__ W,
                                 unsigned short* __restrict__ out) {
  int idx = blockIdx.x * blockDim.x + threadIdx.x;   // 4*8*32*16 = 16384
  if (idx >= 16384) return;
  int p    = idx & 15;
  int lane = (idx >> 4) & 31;
  int nt   = (idx >> 9) & 7;
  int kt   = idx >> 12;
  int n  = nt * 16 + (lane & 15);
  int hh = lane >> 4;
  int k  = kt * 32 + (p < 8 ? 8 * hh + p : 16 + 8 * hh + (p - 8));
  out[idx] = f32_to_bf16_rne(W[k * 128 + n]);
}

// -------------------------- edge scatter (mean agg) ------------------------

// One wave32 per edge; lane handles 4 consecutive features (float4 load).
__global__ void scatter_edges(const float* __restrict__ feat,
                              const int* __restrict__ src,
                              const int* __restrict__ dst,
                              float* __restrict__ ssum,
                              float* __restrict__ cnt, int E) {
  int gid = blockIdx.x * blockDim.x + threadIdx.x;
  int e = gid >> 5, lane = gid & 31;
  if (e >= E) return;
  int sn = src[e], dn = dst[e];
  float4 v = reinterpret_cast<const float4*>(feat + (size_t)sn * 128)[lane];
  float* o = ssum + (size_t)dn * 128 + lane * 4;
  atomicAdd(o + 0, v.x); atomicAdd(o + 1, v.y);
  atomicAdd(o + 2, v.z); atomicAdd(o + 3, v.w);
  if (lane == 0) atomicAdd(&cnt[dn], 1.0f);
}

__global__ void agg_finalize(const float* __restrict__ ssum,
                             const float* __restrict__ cnt,
                             unsigned short* __restrict__ aggb, int n) {
  int i = blockIdx.x * blockDim.x + threadIdx.x;
  if (i >= n) return;
  float c = cnt[i >> 7];
  aggb[i] = f32_to_bf16_rne(ssum[i] / fmaxf(c, 1.0f));
}

// -------------------------- WMMA GEMM --------------------------------------
// out[r,:] = act( A1 @ B1 (+ A2 @ B2) + bias ), all 128 cols.
// Block = 256 threads = 8 waves; wave w owns 16-col tile n0 = 16w;
// blockIdx.x owns a 16-row tile. A tiles staged in LDS once (rows padded to
// 272B = 68 banks so the 16-row stride rotates across banks). K = 128 ->
// 4 wmma steps per operand. ACT: 0=none 1=relu 2=tanh.
static constexpr int APAD = 136;   // ushorts per LDS row (128 + 8 pad)

template <int ACT, bool DUAL>
__global__ void wmma_gemm128(const unsigned short* __restrict__ A1,
                             const unsigned short* __restrict__ B1p,
                             const unsigned short* __restrict__ A2,
                             const unsigned short* __restrict__ B2p,
                             const float* __restrict__ bias,
                             float* __restrict__ outF,
                             unsigned short* __restrict__ outB,
                             int nrows) {
  __shared__ unsigned short sA[(DUAL ? 2 : 1) * 16 * APAD];
  const int tid  = threadIdx.x;
  const int lane = tid & 31;
  const int wave = tid >> 5;
  const int m0 = blockIdx.x * 16;
  if (m0 >= nrows) return;
  const int n0 = wave * 16;
  const int h  = lane >> 4;
  const int ml = lane & 15;
  const int nn = n0 + ml;

  // Cooperative stage of A tile(s): 16 rows x 128 bf16 each (16B chunks).
  {
    const int nseg = DUAL ? 512 : 256;       // uint4 chunks total
    for (int idx = tid; idx < nseg; idx += 256) {
      const int mat = idx >> 8;
      const int row = (idx >> 4) & 15;
      const int seg = idx & 15;
      const unsigned short* g = (mat ? A2 : A1) + (size_t)(m0 + row) * 128 + seg * 8;
      *reinterpret_cast<uint4*>(&sA[mat * (16 * APAD) + row * APAD + seg * 8]) =
          *reinterpret_cast<const uint4*>(g);
    }
  }
  __syncthreads();

  v8f c;
  const float bv = bias ? bias[nn] : 0.0f;
#pragma unroll
  for (int r = 0; r < 8; ++r) c[r] = bv;

  const unsigned short* s1 = &sA[ml * APAD];
  const unsigned short* s2 = &sA[(DUAL ? 16 * APAD : 0) + ml * APAD];

#pragma unroll
  for (int kt = 0; kt < 4; ++kt) {
    const int kb = kt * 32;
    FragBF a, b;
    // A 16-bit layout: lanes h=0 take K = kb+[0..7] & kb+16+[0..7]; h=1 shifts by 8.
    a.u[0] = *reinterpret_cast<const uint4*>(s1 + kb + 8 * h);
    a.u[1] = *reinterpret_cast<const uint4*>(s1 + kb + 16 + 8 * h);
    const uint4* bp =
        reinterpret_cast<const uint4*>(B1p + ((size_t)(kt * 8 + wave) * 512) + lane * 16);
    b.u[0] = bp[0]; b.u[1] = bp[1];
    c = __builtin_amdgcn_wmma_f32_16x16x32_bf16(false, a.v, false, b.v,
                                                (short)0, c, false, false);
    if (DUAL) {
      FragBF a2, b2;
      a2.u[0] = *reinterpret_cast<const uint4*>(s2 + kb + 8 * h);
      a2.u[1] = *reinterpret_cast<const uint4*>(s2 + kb + 16 + 8 * h);
      const uint4* b2p =
          reinterpret_cast<const uint4*>(B2p + ((size_t)(kt * 8 + wave) * 512) + lane * 16);
      b2.u[0] = b2p[0]; b2.u[1] = b2p[1];
      c = __builtin_amdgcn_wmma_f32_16x16x32_bf16(false, a2.v, false, b2.v,
                                                  (short)0, c, false, false);
    }
  }

#pragma unroll
  for (int r = 0; r < 8; ++r) {
    float v = c[r];
    if (ACT == 1) v = fmaxf(v, 0.0f);
    else if (ACT == 2) v = tanhf(v);
    const size_t o = (size_t)(m0 + r + 8 * h) * 128 + nn;  // C layout: M = r + 8h, N = nn
    outF[o] = v;
    if (outB) outB[o] = f32_to_bf16_rne(v);
  }
}

// -------------------------- attention --------------------------------------

// a = softmax(t @ Wc2 + bc2) over 2 cols; one wave per node.
__global__ void attn_softmax2(const float* __restrict__ t,
                              const float* __restrict__ Wc2,
                              const float* __restrict__ bc2,
                              float* __restrict__ a, int N) {
  int gid = blockIdx.x * blockDim.x + threadIdx.x;
  int node = gid >> 5, lane = gid & 31;
  if (node >= N) return;
  float4 v = reinterpret_cast<const float4*>(t + (size_t)node * 128)[lane];
  int k0 = lane * 4;
  float p0 = v.x * Wc2[k0 * 2]       + v.y * Wc2[(k0 + 1) * 2] +
             v.z * Wc2[(k0 + 2) * 2] + v.w * Wc2[(k0 + 3) * 2];
  float p1 = v.x * Wc2[k0 * 2 + 1]       + v.y * Wc2[(k0 + 1) * 2 + 1] +
             v.z * Wc2[(k0 + 2) * 2 + 1] + v.w * Wc2[(k0 + 3) * 2 + 1];
  for (int off = 16; off; off >>= 1) {
    p0 += __shfl_down(p0, off, 32);
    p1 += __shfl_down(p1, off, 32);
  }
  if (lane == 0) {
    float l0 = p0 + bc2[0], l1 = p1 + bc2[1];
    float m = fmaxf(l0, l1);
    float e0 = __expf(l0 - m), e1 = __expf(l1 - m);
    float inv = 1.0f / (e0 + e1);
    a[node * 2 + 0] = e0 * inv;
    a[node * 2 + 1] = e1 * inv;
  }
}

// pos[batch[i],:] += a[i,0] * h[i,:]  (wave per node)
__global__ void pos_scatter(const float* __restrict__ h,
                            const float* __restrict__ a,
                            const int* __restrict__ batch,
                            float* __restrict__ pos, int N) {
  int gid = blockIdx.x * blockDim.x + threadIdx.x;
  int node = gid >> 5, lane = gid & 31;
  if (node >= N) return;
  float pa = a[node * 2];
  int g = batch[node];
  float4 v = reinterpret_cast<const float4*>(h + (size_t)node * 128)[lane];
  float* o = pos + (size_t)g * 128 + lane * 4;
  atomicAdd(o + 0, pa * v.x); atomicAdd(o + 1, pa * v.y);
  atomicAdd(o + 2, pa * v.z); atomicAdd(o + 3, pa * v.w);
}

// column sums of h (N x 128) -> gsum[128]
__global__ void col_sum(const float* __restrict__ h, float* __restrict__ gsum, int N) {
  int c = threadIdx.x;              // 128 threads
  int r0 = blockIdx.x * 128;
  int r1 = min(r0 + 128, N);
  float s = 0.0f;
  for (int r = r0; r < r1; ++r) s += h[(size_t)r * 128 + c];
  atomicAdd(&gsum[c], s);
}

__global__ void broadcast_emb(const float* __restrict__ gsum,
                              float* __restrict__ emb, int total, float invN) {
  int i = blockIdx.x * blockDim.x + threadIdx.x;
  if (i < total) emb[i] = gsum[i & 127] * invN;
}

// new_adj accumulation: per edge 2x2 outer product if same graph
__global__ void adj_scatter(const int* __restrict__ src, const int* __restrict__ dst,
                            const int* __restrict__ batch, const float* __restrict__ a,
                            float* __restrict__ adj, int E) {
  int e = blockIdx.x * blockDim.x + threadIdx.x;
  if (e >= E) return;
  int s = src[e], d = dst[e];
  int g1 = batch[s], g2 = batch[d];
  if (g1 != g2) return;
  float as0 = a[s * 2], as1 = a[s * 2 + 1];
  float ad0 = a[d * 2], ad1 = a[d * 2 + 1];
  float* o = adj + g1 * 4;
  atomicAdd(o + 0, as0 * ad0); atomicAdd(o + 1, as0 * ad1);
  atomicAdd(o + 2, as1 * ad0); atomicAdd(o + 3, as1 * ad1);
}

__global__ void penalty_reduce(const float* __restrict__ adj, float* __restrict__ out, int G) {
  __shared__ float sm[512];
  int g = threadIdx.x;              // 512 threads, one block
  float val = 0.0f;
  if (g < G) {
    float a00 = adj[g * 4], a01 = adj[g * 4 + 1];
    float a10 = adj[g * 4 + 2], a11 = adj[g * 4 + 3];
    float l0 = fabsf(a00) + fabsf(a01);
    float l1 = fabsf(a10) + fabsf(a11);
    float d0 = a00 / fmaxf(l0, 1e-5f);
    float d1 = a11 / fmaxf(l1, 1e-5f);
    val = 0.5f * ((d0 - 1.0f) * (d0 - 1.0f) + (d1 - 1.0f) * (d1 - 1.0f));
  }
  sm[g] = val;
  __syncthreads();
  for (int s = 256; s; s >>= 1) {
    if (g < s) sm[g] += sm[g + s];
    __syncthreads();
  }
  if (g == 0) out[0] = sm[0] / (float)G;
}

// logits + log_softmax, one thread per graph
__global__ void head_logits(const float* __restrict__ z, const float* __restrict__ Wl2,
                            const float* __restrict__ bl2, float* __restrict__ out, int G) {
  int g = blockIdx.x * blockDim.x + threadIdx.x;
  if (g >= G) return;
  float acc[Cc];
#pragma unroll
  for (int c = 0; c < Cc; ++c) acc[c] = bl2[c];
  const float* zr = z + (size_t)g * 128;
  for (int k = 0; k < 128; ++k) {
    float zv = zr[k];
#pragma unroll
    for (int c = 0; c < Cc; ++c) acc[c] += zv * Wl2[k * Cc + c];
  }
  float m = acc[0];
#pragma unroll
  for (int c = 1; c < Cc; ++c) m = fmaxf(m, acc[c]);
  float s = 0.0f;
#pragma unroll
  for (int c = 0; c < Cc; ++c) s += __expf(acc[c] - m);
  float lse = logf(s);
#pragma unroll
  for (int c = 0; c < Cc; ++c) out[g * Cc + c] = acc[c] - m - lse;
}

// ---------------------------------------------------------------------------

static inline size_t alignUp(size_t v) { return (v + 255) & ~size_t(255); }

extern "C" void kernel_launch(void* const* d_in, const int* in_sizes, int n_in,
                              void* d_out, int out_size, void* d_ws, size_t ws_size,
                              hipStream_t stream) {
  (void)n_in; (void)out_size; (void)ws_size; (void)in_sizes;
  const float* x    = (const float*)d_in[0];
  const int*   eidx = (const int*)d_in[1];
  const int*   src  = eidx;
  const int*   dst  = eidx + Ee;
  const int*   batch = (const int*)d_in[2];
  const float* W1l = (const float*)d_in[4];
  const float* b1l = (const float*)d_in[5];
  const float* W1r = (const float*)d_in[6];
  const float* W2l = (const float*)d_in[7];
  const float* b2l = (const float*)d_in[8];
  const float* W2r = (const float*)d_in[9];
  const float* Wc1 = (const float*)d_in[10];
  const float* bc1 = (const float*)d_in[11];
  const float* Wc2 = (const float*)d_in[12];
  const float* bc2 = (const float*)d_in[13];
  const float* Wl1 = (const float*)d_in[14];
  const float* bl1 = (const float*)d_in[15];
  const float* Wl2 = (const float*)d_in[16];
  const float* bl2 = (const float*)d_in[17];

  float* out        = (float*)d_out;
  float* out_logits = out;                         // G*C
  float* out_pos    = out + Gg * Cc;               // G*H
  float* out_emb    = out + Gg * Cc + Gg * Hh;     // G*H
  float* out_pen    = out + Gg * Cc + 2 * Gg * Hh; // 1

  // ---- workspace arena ----
  char* ws = (char*)d_ws;
  size_t off = 0;
  auto take = [&](size_t bytes) { char* p = ws + off; off = alignUp(off + bytes); return p; };
  unsigned short* xb    = (unsigned short*)take((size_t)Nn * 128 * 2); // x  bf16; reused as h2 bf16
  unsigned short* aggb  = (unsigned short*)take((size_t)Nn * 128 * 2); // agg bf16
  unsigned short* hb    = (unsigned short*)take((size_t)Nn * 128 * 2); // h1 bf16
  float* ssum           = (float*)take((size_t)Nn * 128 * 4);          // seg-sum; reused as t
  float* h1             = (float*)take((size_t)Nn * 128 * 4);
  float* h2             = (float*)take((size_t)Nn * 128 * 4);
  float* av             = (float*)take((size_t)Nn * 2 * 4);
  float* cnt            = (float*)take((size_t)Nn * 4);
  unsigned short* W1lp  = (unsigned short*)take(16384 * 2);
  unsigned short* W1rp  = (unsigned short*)take(16384 * 2);
  unsigned short* W2lp  = (unsigned short*)take(16384 * 2);
  unsigned short* W2rp  = (unsigned short*)take(16384 * 2);
  unsigned short* Wc1p  = (unsigned short*)take(16384 * 2);
  unsigned short* Wl1p  = (unsigned short*)take(16384 * 2);
  unsigned short* posb  = (unsigned short*)take((size_t)Gg * 128 * 2);
  float* zbuf           = (float*)take((size_t)Gg * 128 * 4);
  float* gsum           = (float*)take(128 * 4);
  float* adj            = (float*)take((size_t)Gg * 4 * 4);

  const int NF = Nn * 128;
  const int TB = 256;

  // ---- prep: casts + weight packing ----
  hipLaunchKernelGGL(cast_f32_to_bf16, dim3((NF + TB - 1) / TB), dim3(TB), 0, stream, x, xb, NF);
  hipLaunchKernelGGL(pack_weight_bf16, dim3(64), dim3(TB), 0, stream, W1l, W1lp);
  hipLaunchKernelGGL(pack_weight_bf16, dim3(64), dim3(TB), 0, stream, W1r, W1rp);
  hipLaunchKernelGGL(pack_weight_bf16, dim3(64), dim3(TB), 0, stream, W2l, W2lp);
  hipLaunchKernelGGL(pack_weight_bf16, dim3(64), dim3(TB), 0, stream, W2r, W2rp);
  hipLaunchKernelGGL(pack_weight_bf16, dim3(64), dim3(TB), 0, stream, Wc1, Wc1p);
  hipLaunchKernelGGL(pack_weight_bf16, dim3(64), dim3(TB), 0, stream, Wl1, Wl1p);

  // ---- layer 1: mean-agg + dual GEMM + relu ----
  hipMemsetAsync(ssum, 0, (size_t)NF * 4, stream);
  hipMemsetAsync(cnt, 0, (size_t)Nn * 4, stream);
  hipLaunchKernelGGL(scatter_edges, dim3(Ee * 32 / TB), dim3(TB), 0, stream, x, src, dst, ssum, cnt, Ee);
  hipLaunchKernelGGL(agg_finalize, dim3((NF + TB - 1) / TB), dim3(TB), 0, stream, ssum, cnt, aggb, NF);
  hipLaunchKernelGGL((wmma_gemm128<1, true>), dim3(Nn / 16), dim3(TB), 0, stream,
                     aggb, W1lp, xb, W1rp, b1l, h1, hb, Nn);

  // ---- layer 2 ----
  hipMemsetAsync(ssum, 0, (size_t)NF * 4, stream);
  hipMemsetAsync(cnt, 0, (size_t)Nn * 4, stream);
  hipLaunchKernelGGL(scatter_edges, dim3(Ee * 32 / TB), dim3(TB), 0, stream, h1, src, dst, ssum, cnt, Ee);
  hipLaunchKernelGGL(agg_finalize, dim3((NF + TB - 1) / TB), dim3(TB), 0, stream, ssum, cnt, aggb, NF);
  hipLaunchKernelGGL((wmma_gemm128<1, true>), dim3(Nn / 16), dim3(TB), 0, stream,
                     aggb, W2lp, hb, W2rp, b2l, h2, xb /*h2 bf16*/, Nn);

  // ---- attention: t = tanh(h2 @ Wc1 + bc1); a = softmax(t @ Wc2 + bc2) ----
  hipLaunchKernelGGL((wmma_gemm128<2, false>), dim3(Nn / 16), dim3(TB), 0, stream,
                     xb, Wc1p, nullptr, nullptr, bc1, ssum /*t*/, nullptr, Nn);
  hipLaunchKernelGGL(attn_softmax2, dim3(Nn * 32 / TB), dim3(TB), 0, stream, ssum, Wc2, bc2, av, Nn);

  // ---- pooling: pos, graph_emb, adjacency penalty ----
  hipMemsetAsync(out_pos, 0, (size_t)Gg * 128 * 4, stream);
  hipMemsetAsync(gsum, 0, 128 * 4, stream);
  hipMemsetAsync(adj, 0, (size_t)Gg * 4 * 4, stream);
  hipLaunchKernelGGL(pos_scatter, dim3(Nn * 32 / TB), dim3(TB), 0, stream, h2, av, batch, out_pos, Nn);
  hipLaunchKernelGGL(col_sum, dim3((Nn + 127) / 128), dim3(128), 0, stream, h2, gsum, Nn);
  hipLaunchKernelGGL(broadcast_emb, dim3(Gg * 128 / TB), dim3(TB), 0, stream,
                     gsum, out_emb, Gg * 128, 1.0f / (float)Nn);
  hipLaunchKernelGGL(adj_scatter, dim3((Ee + TB - 1) / TB), dim3(TB), 0, stream, src, dst, batch, av, adj, Ee);
  hipLaunchKernelGGL(penalty_reduce, dim3(1), dim3(512), 0, stream, adj, out_pen, Gg);

  // ---- head: z = relu(pos @ Wl1 + bl1); log_softmax(z @ Wl2 + bl2) ----
  hipLaunchKernelGGL(cast_f32_to_bf16, dim3(Gg * 128 / TB), dim3(TB), 0, stream, out_pos, posb, Gg * 128);
  hipLaunchKernelGGL((wmma_gemm128<1, false>), dim3(Gg / 16), dim3(TB), 0, stream,
                     posb, Wl1p, nullptr, nullptr, bl1, zbuf, nullptr, Gg);
  hipLaunchKernelGGL(head_logits, dim3((Gg + TB - 1) / TB), dim3(TB), 0, stream, zbuf, Wl2, bl2, out_logits, Gg);
}
